// Lstm_84705345011938
// MI455X (gfx1250) — matrix operand order repaired and checked by reading, not compiled, not run on hardware
//
#include <hip/hip_runtime.h>

// LSTM forward for MI455X (gfx1250, wave32, WMMA).
// B=64, S=512, D=1024, H=1024, gates order: i, f, o, c_hat.
//
//  - One-time fp32 -> bf16 conversion of x, Wx, Wh into workspace.
//  - 512 sequential fused step kernels: g = [x_t | h] @ [Wx | Wh]^T + b,
//    activations, cell update; h written fp32 (output) + bf16 (next step).
//  - Weights (16 MB bf16) stay L2-resident across all steps (192 MB L2).
//  - A-panel [x_t | h] (64 x 2048 bf16, 256 KB) staged ONCE per block into
//    LDS (320 KB/WGP) -> WMMA A-fragments come from low-latency ds_load_b128,
//    and the 4x redundant global A reads across gates are eliminated.
//  - Weight B-fragments: register double-buffer + prefetch (sequential,
//    L2-resident stream; only 2 loads per chunk per wave).
//  - WMMA: v_wmma_f32_16x16x32_bf16, fp32 accumulation.

typedef __attribute__((ext_vector_type(16))) __bf16 v16bf;
typedef __attribute__((ext_vector_type(8)))  float  v8f;

constexpr int LB = 64;    // batch
constexpr int LS = 512;   // seq len
constexpr int LD = 1024;  // input dim
constexpr int LH = 1024;  // hidden dim
constexpr int KTOT = LD + LH;          // 2048 combined K
constexpr int APITCH = KTOT + 8;       // 2056 bf16: 4-bank skew per row

__device__ __forceinline__ unsigned short f32_to_bf16_bits(float f) {
  unsigned int u = __float_as_uint(f);
  u += 0x7FFFu + ((u >> 16) & 1u);   // round-to-nearest-even
  return (unsigned short)(u >> 16);
}

union FragU {
  v16bf v;
  uint4 q[2];
};

// B-matrix fragment (32x16 bf16) from row-major W (N x K), i.e. B = W^T.
// Lanes 0-15: column N=lane, K 0..15; lanes 16-31: column N=lane-16, K 16..31.
__device__ __forceinline__ v16bf load_b_frag(const unsigned short* base,
                                             size_t rowStride, int lane) {
  int n    = lane & 15;
  int koff = (lane >> 4) << 4;      // 0 or 16
  const unsigned short* p = base + (size_t)n * rowStride + koff;
  FragU f;
  f.q[0] = *reinterpret_cast<const uint4*>(p);        // K = koff .. koff+7
  f.q[1] = *reinterpret_cast<const uint4*>(p + 8);    // K = koff+8 .. koff+15
  return f.v;
}

// A-matrix fragment (16x32 bf16) from the LDS A-panel.
// Lanes 0-15 hold row M=lane, K 0..15;
// lanes 16-31 hold row M=lane-16, K 16..31 (matches B fragment order).
__device__ __forceinline__ v16bf lds_a_frag(const unsigned short* abuf,
                                            int mbase, int kk, int lane) {
  int r  = mbase + (lane & 15);
  int kh = (lane >> 4) << 4;        // 0 or 16
  const unsigned short* p = abuf + r * APITCH + kk + kh;
  FragU f;
  f.q[0] = *reinterpret_cast<const uint4*>(p);        // K = kh .. kh+7
  f.q[1] = *reinterpret_cast<const uint4*>(p + 8);    // K = kh+8 .. kh+15
  return f.v;
}

__device__ __forceinline__ v8f wmma_bf16(v16bf a, v16bf b, v8f c) {
  return __builtin_amdgcn_wmma_f32_16x16x32_bf16(false, a, false, b,
                                                 (short)0, c, false, false);
}

__global__ __launch_bounds__(256) void convert_f32_to_bf16(
    const float* __restrict__ src, unsigned short* __restrict__ dst, long n4) {
  long i = (long)blockIdx.x * blockDim.x + threadIdx.x;
  if (i >= n4) return;
  float4 v = reinterpret_cast<const float4*>(src)[i];
  ushort4 o;
  o.x = f32_to_bf16_bits(v.x);
  o.y = f32_to_bf16_bits(v.y);
  o.z = f32_to_bf16_bits(v.z);
  o.w = f32_to_bf16_bits(v.w);
  reinterpret_cast<ushort4*>(dst)[i] = o;
}

__global__ __launch_bounds__(256) void init_state(
    unsigned short* __restrict__ h16a, unsigned short* __restrict__ h16b,
    float* __restrict__ c, int n) {
  int i = blockIdx.x * blockDim.x + threadIdx.x;
  if (i >= n) return;
  h16a[i] = 0;
  h16b[i] = 0;
  c[i]    = 0.0f;
}

// One fused timestep. Grid: H/16 = 64 blocks, 256 threads (8 waves).
// Block owns columns [n0, n0+16) of H for ALL 4 gates (fused activations).
// Wave w: gate = w&3, M-tiles (w>>2) and (w>>2)+2  (M = batch = 64 -> 4 tiles).
__global__ __launch_bounds__(256) void lstm_step_kernel(
    const unsigned short* __restrict__ x16,   // (B, S, D) bf16
    const unsigned short* __restrict__ wx16,  // (4, H, D) bf16
    const unsigned short* __restrict__ wh16,  // (4, H, H) bf16
    const float* __restrict__ bias,           // (4, H) fp32
    const unsigned short* __restrict__ h_in,  // (B, H) bf16
    unsigned short* __restrict__ h_out,       // (B, H) bf16
    float* __restrict__ cstate,               // (B, H) fp32, in-place
    float* __restrict__ out_y,                // (B, S, H) fp32
    float* __restrict__ h_last,               // (B, H) fp32
    float* __restrict__ c_last,               // (B, H) fp32
    int t, int is_last) {
  __shared__ unsigned short abuf[LB * APITCH];  // A-panel [x_t | h], 263168 B
  __shared__ float gbuf[4 * 64 * 17];           // gate exchange, 17408 B

  const int tid  = threadIdx.x;
  const int lane = tid & 31;
  const int wave = tid >> 5;
  const int gate = wave & 3;
  const int mt0  = wave >> 2;                 // M-tiles mt0 and mt0+2
  const int n0   = blockIdx.x * 16;

  // ---- Stage A-panel into LDS: 64 rows x (1024 x | 1024 h) bf16 ----
  // 16-byte segments; x part: 64 rows * 128 segs, h part: same.
  for (int s = tid; s < LB * (LD / 8); s += 256) {
    int row = s >> 7;                 // LD/8 = 128 segs per row
    int seg = s & 127;
    uint4 v = *reinterpret_cast<const uint4*>(
        x16 + ((size_t)row * LS + t) * LD + seg * 8);
    *reinterpret_cast<uint4*>(abuf + row * APITCH + seg * 8) = v;
  }
  for (int s = tid; s < LB * (LH / 8); s += 256) {
    int row = s >> 7;
    int seg = s & 127;
    uint4 v = *reinterpret_cast<const uint4*>(
        h_in + (size_t)row * LH + seg * 8);
    *reinterpret_cast<uint4*>(abuf + row * APITCH + LD + seg * 8) = v;
  }
  __syncthreads();

  v8f acc0 = {};
  v8f acc1 = {};

  const unsigned short* wxg = wx16 + ((size_t)gate * LH + n0) * LD;
  const unsigned short* whg = wh16 + ((size_t)gate * LH + n0) * LH;
  const int m0 = mt0 * 16;
  const int m1 = (mt0 + 2) * 16;

  // ---- K-loop: A from LDS, weight B-fragments register double-buffered ----
  v16bf bf_c = load_b_frag(wxg, LD, lane);
  v16bf bf_n = load_b_frag(wxg + 32, LD, lane);

  // Phase A: K = 0 .. D  (weights from Wx)
#pragma unroll 4
  for (int kk = 0; kk < LD; kk += 32) {
    __builtin_prefetch(wxg + kk + 128, 0, 1);
    v16bf a0 = lds_a_frag(abuf, m0, kk, lane);
    v16bf a1 = lds_a_frag(abuf, m1, kk, lane);
    acc0 = wmma_bf16(a0, bf_c, acc0);
    acc1 = wmma_bf16(a1, bf_c, acc1);
    bf_c = bf_n;
    int kn = kk + 64;
    bf_n = (kn < LD) ? load_b_frag(wxg + kn, LD, lane)
                     : load_b_frag(whg + (kn - LD), LH, lane);
  }
  // Phase B: K = D .. D+H  (weights from Wh)
#pragma unroll 4
  for (int kk = 0; kk < LH; kk += 32) {
    __builtin_prefetch(whg + kk + 128, 0, 1);
    v16bf a0 = lds_a_frag(abuf, m0, LD + kk, lane);
    v16bf a1 = lds_a_frag(abuf, m1, LD + kk, lane);
    acc0 = wmma_bf16(a0, bf_c, acc0);
    acc1 = wmma_bf16(a1, bf_c, acc1);
    bf_c = bf_n;
    int kn = kk + 64;
    if (kn < LH) bf_n = load_b_frag(whg + kn, LH, lane);
  }

  // C/D layout (ISA 7.12.2): VGPR r, lanes 0-15 -> M=r, N=lane;
  // lanes 16-31 -> M=8+r, N=lane-16.
  {
    const int mrow_off = (lane >> 4) ? 8 : 0;
    const int n   = lane & 15;
    const int mm0 = m0 + mrow_off;
    const int mm1 = m1 + mrow_off;
#pragma unroll
    for (int r = 0; r < 8; ++r) {
      gbuf[(gate * 64 + mm0 + r) * 17 + n] = acc0[r];
      gbuf[(gate * 64 + mm1 + r) * 17 + n] = acc1[r];
    }
  }
  __syncthreads();

  // Fused epilogue: bias + gates + cell update. 1024 elements / 256 threads.
  for (int e = tid; e < 64 * 16; e += 256) {
    const int m = e >> 4;      // batch row
    const int n = e & 15;      // column within [n0, n0+16)
    const int col = n0 + n;

    float gi = gbuf[(0 * 64 + m) * 17 + n] + bias[0 * LH + col];
    float gf = gbuf[(1 * 64 + m) * 17 + n] + bias[1 * LH + col];
    float go = gbuf[(2 * 64 + m) * 17 + n] + bias[2 * LH + col];
    float gc = gbuf[(3 * 64 + m) * 17 + n] + bias[3 * LH + col];

    float it = 1.0f / (1.0f + __expf(-gi));
    float ft = 1.0f / (1.0f + __expf(-gf));
    float ot = 1.0f / (1.0f + __expf(-go));
    float ch = tanhf(gc);

    const size_t cidx = (size_t)m * LH + col;
    float cn = ft * cstate[cidx] + it * ch;
    float hn = ot * tanhf(cn);

    cstate[cidx] = cn;
    out_y[((size_t)m * LS + t) * LH + col] = hn;
    h_out[cidx] = f32_to_bf16_bits(hn);
    if (is_last) {
      h_last[cidx] = hn;
      c_last[cidx] = cn;
    }
  }
}

extern "C" void kernel_launch(void* const* d_in, const int* in_sizes, int n_in,
                              void* d_out, int out_size, void* d_ws, size_t ws_size,
                              hipStream_t stream) {
  (void)in_sizes; (void)n_in; (void)out_size; (void)ws_size;

  const float* x    = (const float*)d_in[0];  // (B, S, D)
  const float* Wx   = (const float*)d_in[1];  // (4, H, D)
  const float* Wh   = (const float*)d_in[2];  // (4, H, H)
  const float* bias = (const float*)d_in[3];  // (4, H)

  float* out_y  = (float*)d_out;                       // (B, S, H)
  float* h_last = out_y + (size_t)LB * LS * LH;        // (B, H)
  float* c_last = h_last + (size_t)LB * LH;            // (B, H)

  // Workspace layout (~84.4 MB total)
  unsigned short* x16  = (unsigned short*)d_ws;                 // B*S*D bf16
  unsigned short* wx16 = x16 + (size_t)LB * LS * LD;            // 4*H*D bf16
  unsigned short* wh16 = wx16 + (size_t)4 * LH * LD;            // 4*H*H bf16
  unsigned short* h16a = wh16 + (size_t)4 * LH * LH;            // B*H bf16 (ping)
  unsigned short* h16b = h16a + (size_t)LB * LH;                // B*H bf16 (pong)
  float* cst = (float*)(h16b + (size_t)LB * LH);                // B*H fp32

  // One-time bf16 conversions
  {
    size_t n = (size_t)LB * LS * LD;
    convert_f32_to_bf16<<<(int)((n / 4 + 255) / 256), 256, 0, stream>>>(x, x16, (long)(n / 4));
    n = (size_t)4 * LH * LD;
    convert_f32_to_bf16<<<(int)((n / 4 + 255) / 256), 256, 0, stream>>>(Wx, wx16, (long)(n / 4));
    n = (size_t)4 * LH * LH;
    convert_f32_to_bf16<<<(int)((n / 4 + 255) / 256), 256, 0, stream>>>(Wh, wh16, (long)(n / 4));
  }

  // h0 = 0, c0 = 0
  {
    int n = LB * LH;
    init_state<<<(n + 255) / 256, 256, 0, stream>>>(h16a, h16b, cst, n);
  }

  // Sequential recurrence: one fused kernel per timestep (ping-pong h buffers).
  for (int t = 0; t < LS; ++t) {
    const unsigned short* hin  = (t & 1) ? h16b : h16a;
    unsigned short*       hout = (t & 1) ? h16a : h16b;
    lstm_step_kernel<<<LH / 16, 256, 0, stream>>>(
        x16, wx16, wh16, bias, hin, hout, cst, out_y, h_last, c_last,
        t, (t == LS - 1) ? 1 : 0);
  }
}
